// CustomDenseLayer_74406013436176
// MI455X (gfx1250) — matrix operand order repaired and checked by reading, not compiled, not run on hardware
//
#include <hip/hip_runtime.h>
#include <hip/hip_bf16.h>
#include <stdint.h>

// Problem constants (fixed by the reference).
#define BATCH       1024
#define IN_SZ       2048
#define OUT_SZ      2048

typedef __attribute__((ext_vector_type(16))) __bf16 v16bf;
typedef __attribute__((ext_vector_type(8)))  float  v8f;

union AFrag { uint4 q[2]; v16bf v; };

// ---------------------------------------------------------------------------
// CDNA5 async global->LDS copy (ASYNCcnt path, bypasses VGPRs).
// vdst = LDS byte address (low 32 bits of generic shared pointer),
// vaddr = 64-bit global address, GV mode (saddr = off).
// ---------------------------------------------------------------------------
__device__ __forceinline__ void async_copy_b128(const void* gptr, void* lptr) {
    uint32_t lds = (uint32_t)(uintptr_t)lptr;
    uint64_t ga  = (uint64_t)(uintptr_t)gptr;
    asm volatile("global_load_async_to_lds_b128 %0, %1, off"
                 :: "v"(lds), "v"(ga) : "memory");
}
__device__ __forceinline__ void wait_async0() {
    asm volatile("s_wait_asynccnt 0x0" ::: "memory");
}

// ---------------------------------------------------------------------------
// Kernel 1: zero the fp32 densification buffer Wf[OUT_SZ][IN_SZ]
// ---------------------------------------------------------------------------
__global__ __launch_bounds__(256) void k_zeroW(float4* __restrict__ Wf4) {
    int i = blockIdx.x * blockDim.x + threadIdx.x;   // covers OUT_SZ*IN_SZ/4
    Wf4[i] = make_float4(0.f, 0.f, 0.f, 0.f);
}

// ---------------------------------------------------------------------------
// Kernel 2: scatter edges into TRANSPOSED dense W: Wf[out][in] += w
// ---------------------------------------------------------------------------
__global__ __launch_bounds__(256) void k_scatter(const float* __restrict__ w,
                                                 const int*   __restrict__ in_idx,
                                                 const int*   __restrict__ out_idx,
                                                 float* __restrict__ Wf, int E) {
    int e = blockIdx.x * blockDim.x + threadIdx.x;
    if (e < E) {
        int i = in_idx[e];
        int o = out_idx[e];
        atomicAdd(&Wf[(size_t)o * IN_SZ + i], w[e]);
    }
}

// ---------------------------------------------------------------------------
// Kernel 3: fp32 -> bf16 pack (used for both W and X)
// ---------------------------------------------------------------------------
__global__ __launch_bounds__(256) void k_cvt(const float4* __restrict__ src,
                                             uint2* __restrict__ dst) {
    int i = blockIdx.x * blockDim.x + threadIdx.x;   // 4 elements per thread
    float4 f = src[i];
    union { __bf16 h[4]; uint2 u; } pk;
    pk.h[0] = (__bf16)f.x;
    pk.h[1] = (__bf16)f.y;
    pk.h[2] = (__bf16)f.z;
    pk.h[3] = (__bf16)f.w;
    dst[i] = pk.u;
}

// ---------------------------------------------------------------------------
// Kernel 4: GEMM  out[M=1024][N=2048] = Xb[M][K=2048] * Wt[N][K]^T
// 256 threads = 8 wave32, block tile 128x128, K-block 64, double-buffered LDS
// filled by GLOBAL_LOAD_ASYNC_TO_LDS_B128. Wave grid 2x4: each wave owns
// 64x32 of C as 4x2 accumulators of 16x16 (v_wmma_f32_16x16x32_bf16).
// ---------------------------------------------------------------------------
#define LDK 72            // padded LDS k-stride (bf16 elements), 144B rows
#define KBLK 64
#define NKITER (IN_SZ / KBLK)   // 32

__global__ __launch_bounds__(256) void k_gemm(const __bf16* __restrict__ Xb,
                                              const __bf16* __restrict__ Wt,
                                              float* __restrict__ out) {
    // [buffer][A=0/B=1][128*LDK] : 2*2*128*72*2B = 72 KB (WGP has 320 KB)
    __shared__ __align__(16) __bf16 smem[2][2][128 * LDK];

    const int tid   = threadIdx.x;
    const int lane  = tid & 31;
    const int wave  = tid >> 5;          // 0..7
    const int mw    = wave >> 2;         // 0..1 -> 64-row slab
    const int nw    = wave & 3;          // 0..3 -> 32-col slab
    const int lr    = lane & 15;
    const int lhalf = lane >> 4;         // 0/1 (K sub-chunk select)

    const int mBlock = blockIdx.y * 128; // 8 blocks over M=1024
    const int nBlock = blockIdx.x * 128; // 16 blocks over N=2048

    // Per-thread staging coordinates: 1024 16B-chunks per 128x64 tile,
    // 4 chunks per thread.
    int srow[4], scch[4];
#pragma unroll
    for (int it = 0; it < 4; ++it) {
        int lin  = tid + it * 256;       // 0..1023
        srow[it] = lin >> 3;             // 0..127
        scch[it] = lin & 7;              // 16B chunk within 64-wide row
    }

    v8f acc[4][2];
#pragma unroll
    for (int i = 0; i < 4; ++i)
#pragma unroll
        for (int j = 0; j < 2; ++j)
            acc[i][j] = (v8f){0.f, 0.f, 0.f, 0.f, 0.f, 0.f, 0.f, 0.f};

    // ---- prologue: async-stage K-block 0 into buffer 0 ----
#pragma unroll
    for (int it = 0; it < 4; ++it) {
        int r = srow[it], c = scch[it] * 8;
        async_copy_b128(Xb + (size_t)(mBlock + r) * IN_SZ + c, &smem[0][0][r * LDK + c]);
        async_copy_b128(Wt + (size_t)(nBlock + r) * IN_SZ + c, &smem[0][1][r * LDK + c]);
    }
    wait_async0();
    __syncthreads();

    for (int i = 0; i < NKITER; ++i) {
        const int p  = i & 1;
        const int kc = i * KBLK;

        // ---- async-stage next K-block into the other buffer ----
        if (i + 1 < NKITER) {
#pragma unroll
            for (int it = 0; it < 4; ++it) {
                int r = srow[it], c = scch[it] * 8;
                async_copy_b128(Xb + (size_t)(mBlock + r) * IN_SZ + (kc + KBLK) + c,
                                &smem[1 - p][0][r * LDK + c]);
                async_copy_b128(Wt + (size_t)(nBlock + r) * IN_SZ + (kc + KBLK) + c,
                                &smem[1 - p][1][r * LDK + c]);
            }
        }

        // ---- compute on current buffer: 2 WMMA K-steps of 32 ----
        const __bf16* As = smem[p][0];
        const __bf16* Bs = smem[p][1];
#pragma unroll
        for (int k0 = 0; k0 < KBLK; k0 += 32) {
            AFrag a[4], b[2];
            // A frag (16x32 bf16, ISA layout): lanes0-15 K=k0..k0+7 &
            // k0+16..k0+23; lanes16-31 shifted by 8.
            const int kbA = k0 + lhalf * 8;
#pragma unroll
            for (int t = 0; t < 4; ++t) {
                int r = mw * 64 + t * 16 + lr;
                a[t].q[0] = *(const uint4*)(&As[r * LDK + kbA]);
                a[t].q[1] = *(const uint4*)(&As[r * LDK + kbA + 16]);
            }
            // B frag (32x16 bf16): lane holds column N=lr, same K order as A
            // (K=k0..k0+7 & k0+16..k0+23; lanes16-31 shifted by 8).
            const int kbB = k0 + lhalf * 8;
#pragma unroll
            for (int t = 0; t < 2; ++t) {
                int n = nw * 32 + t * 16 + lr;
                b[t].q[0] = *(const uint4*)(&Bs[n * LDK + kbB]);
                b[t].q[1] = *(const uint4*)(&Bs[n * LDK + kbB + 16]);
            }
#pragma unroll
            for (int ti = 0; ti < 4; ++ti)
#pragma unroll
                for (int tj = 0; tj < 2; ++tj)
                    acc[ti][tj] = __builtin_amdgcn_wmma_f32_16x16x32_bf16(
                        false, a[ti].v, false, b[tj].v,
                        (short)0, acc[ti][tj], false, false);
        }

        // ---- close the pipeline stage: my async writes done + all waves
        //      finished reading buffer p ----
        wait_async0();
        __syncthreads();
    }

    // ---- epilogue: 32-bit C/D layout: N = lr, M = v + 8*lhalf ----
#pragma unroll
    for (int i = 0; i < 4; ++i) {
#pragma unroll
        for (int j = 0; j < 2; ++j) {
            int n  = nBlock + nw * 32 + j * 16 + lr;
            int m0 = mBlock + mw * 64 + i * 16 + lhalf * 8;
            float* po = out + (size_t)m0 * OUT_SZ + n;
#pragma unroll
            for (int v = 0; v < 8; ++v)
                po[(size_t)v * OUT_SZ] = acc[i][j][v];
        }
    }
}

// ---------------------------------------------------------------------------
// Host-side launch. d_in: x(f32 B*IN), weights(f32 E), in_idx(int E),
// out_idx(int E). d_out: f32 B*OUT. Workspace layout:
//   [0,16MB)  Wf  fp32 dense transposed W accumulation
//   [16,24MB) Wt  bf16 transposed W
//   [24,28MB) Xb  bf16 X
// ---------------------------------------------------------------------------
extern "C" void kernel_launch(void* const* d_in, const int* in_sizes, int n_in,
                              void* d_out, int out_size, void* d_ws, size_t ws_size,
                              hipStream_t stream) {
    const float* x  = (const float*)d_in[0];
    const float* w  = (const float*)d_in[1];
    const int*   ii = (const int*)d_in[2];
    const int*   oi = (const int*)d_in[3];
    const int    E  = in_sizes[1];
    float* out = (float*)d_out;

    char* ws = (char*)d_ws;
    float*  Wf = (float*)ws;
    __bf16* Wt = (__bf16*)(ws + (size_t)16 * 1024 * 1024);
    __bf16* Xb = (__bf16*)(ws + (size_t)24 * 1024 * 1024);

    // 1) zero dense W (fp32)
    k_zeroW<<<(OUT_SZ * IN_SZ / 4) / 256, 256, 0, stream>>>((float4*)Wf);
    // 2) densify (transposed): Wf[out][in] += w
    k_scatter<<<(E + 255) / 256, 256, 0, stream>>>(w, ii, oi, Wf, E);
    // 3) convert W and X to bf16
    k_cvt<<<(OUT_SZ * IN_SZ / 4) / 256, 256, 0, stream>>>((const float4*)Wf, (uint2*)Wt);
    k_cvt<<<(BATCH * IN_SZ / 4) / 256, 256, 0, stream>>>((const float4*)x, (uint2*)Xb);
    // 4) WMMA GEMM with async double-buffered staging: out = Xb @ Wt^T
    dim3 grid(OUT_SZ / 128, BATCH / 128);   // (16, 8)
    k_gemm<<<grid, 256, 0, stream>>>(Xb, Wt, out);
}
